// Block_63247688401352
// MI455X (gfx1250) — matrix operand order repaired and checked by reading, not compiled, not run on hardware
//
#include <hip/hip_runtime.h>
#include <hip/hip_bf16.h>

// ---------------------------------------------------------------------------
// Transformer block (pre-LN, causal MHA + ReLU-FFN) for gfx1250 (MI455X).
// All GEMM-class math uses v_wmma_f32_16x16x32_f16 (f16 in, f32 accum).
// ---------------------------------------------------------------------------

typedef _Float16 v16h __attribute__((ext_vector_type(16)));
typedef _Float16 v8h  __attribute__((ext_vector_type(8)));
typedef float    v8f  __attribute__((ext_vector_type(8)));

#define N_EMBD    1024
#define N_HEADS   16
#define HEAD_SIZE 64
#define SEQ       2048
#define BATCH     2
#define ROWS      (BATCH * SEQ)     // 4096
#define FFN       (4 * N_EMBD)      // 4096

__device__ __forceinline__ v8f zero8() {
    v8f z;
#pragma unroll
    for (int i = 0; i < 8; ++i) z[i] = 0.0f;
    return z;
}

// Build a v16h A/B fragment from two contiguous 8-half (16B) segments.
__device__ __forceinline__ v16h ld_frag_pair(const _Float16* p0, const _Float16* p1) {
    v8h lo = *(const v8h*)p0;
    v8h hi = *(const v8h*)p1;
    v16h r;
#pragma unroll
    for (int i = 0; i < 8; ++i) { r[i] = lo[i]; r[i + 8] = hi[i]; }
    return r;
}

// ---------------------------------------------------------------------------
// f32 -> f16 cast (weights)
// ---------------------------------------------------------------------------
__global__ void cast_f32_to_f16(const float* __restrict__ in, _Float16* __restrict__ out, int n) {
    int i = blockIdx.x * blockDim.x + threadIdx.x;
    int stride = gridDim.x * blockDim.x;
    for (; i < n; i += stride) out[i] = (_Float16)in[i];
}

// ---------------------------------------------------------------------------
// LayerNorm over C=1024, one row per block (256 threads, 4 elems/thread).
// Output is f16 (feeds WMMA A-matrix).
// ---------------------------------------------------------------------------
__global__ void ln_kernel(const float* __restrict__ x, const float* __restrict__ w,
                          const float* __restrict__ b, _Float16* __restrict__ out) {
    const int row  = blockIdx.x;
    const int tid  = threadIdx.x;
    const int lane = tid & 31;
    const int wid  = tid >> 5;

    const float4 v = ((const float4*)(x + (size_t)row * N_EMBD))[tid];
    float s = v.x + v.y + v.z + v.w;
    float q = v.x * v.x + v.y * v.y + v.z * v.z + v.w * v.w;
#pragma unroll
    for (int off = 16; off > 0; off >>= 1) {
        s += __shfl_xor(s, off, 32);
        q += __shfl_xor(q, off, 32);
    }
    __shared__ float rs[8], rq[8];
    if (lane == 0) { rs[wid] = s; rq[wid] = q; }
    __syncthreads();
    float S = 0.f, Q = 0.f;
#pragma unroll
    for (int i = 0; i < 8; ++i) { S += rs[i]; Q += rq[i]; }
    const float mu   = S * (1.0f / N_EMBD);
    const float var  = Q * (1.0f / N_EMBD) - mu * mu;
    const float rstd = rsqrtf(var + 1e-5f);

    const float4 wv = ((const float4*)w)[tid];
    const float4 bv = ((const float4*)b)[tid];
    _Float16 o[4];
    o[0] = (_Float16)((v.x - mu) * rstd * wv.x + bv.x);
    o[1] = (_Float16)((v.y - mu) * rstd * wv.y + bv.y);
    o[2] = (_Float16)((v.z - mu) * rstd * wv.z + bv.z);
    o[3] = (_Float16)((v.w - mu) * rstd * wv.w + bv.w);
    *(uint2*)(out + (size_t)row * N_EMBD + tid * 4) = *(const uint2*)o;
}

// ---------------------------------------------------------------------------
// Tiled WMMA GEMM: out = act(A[M,K](f16) @ W[K,N](f16) + bias) (+ residual)
// Block tile 128x128, K-step 32. 8 waves (2 M x 4 N), each wave 64x32 output
// as 4x2 v_wmma_f32_16x16x32_f16 tiles. LDS double-buffered; global loads for
// step s+1 are issued before the WMMAs of step s so HBM latency hides under
// matrix math. All epilogue behavior is compile-time (no per-element branches).
// OMODE: 0 = f32 row-major, 1 = f16 row-major, 2 = f16 permuted [B,H,T,D].
// ---------------------------------------------------------------------------
template <int N, bool HAS_BIAS, bool HAS_RES, bool RELU, int OMODE>
__global__ __launch_bounds__(256)
void gemm_wmma(const _Float16* __restrict__ A, const _Float16* __restrict__ W,
               const float* __restrict__ bias, const float* __restrict__ residual,
               float* __restrict__ out32, _Float16* __restrict__ out16,
               int M, int K) {
    __shared__ __align__(16) _Float16 As[2][128][40];   // [buf][m][k], pad 8
    __shared__ __align__(16) _Float16 Wt[2][128][40];   // [buf][n][k] (transposed)

    const int tid  = threadIdx.x;
    const int lane = tid & 31;
    const int w    = tid >> 5;
    const int mL   = lane & 15;
    const int half = lane >> 4;

    const int N0 = blockIdx.x * 128;
    const int M0 = blockIdx.y * 128;
    const int wmb = (w >> 2) * 64;   // wave M base within tile
    const int wnb = (w & 3) * 32;    // wave N base within tile

    v8f acc[4][2];
#pragma unroll
    for (int mi = 0; mi < 4; ++mi)
#pragma unroll
        for (int ni = 0; ni < 2; ++ni) acc[mi][ni] = zero8();

    const int arow = tid >> 1, akoff = (tid & 1) * 16;      // A tile loader
    const int wrow = tid >> 3, wnoff = (tid & 7) * 16;      // W tile loader
    const _Float16* gAbase = A + (size_t)(M0 + arow) * K + akoff;
    const _Float16* gWbase = W + (size_t)wrow * N + N0 + wnoff;

    // Prologue: stage step 0 into buffer 0.
    {
        uint4 a0 = *(const uint4*)gAbase;
        uint4 a1 = *(const uint4*)(gAbase + 8);
        v8h  w0 = *(const v8h*)gWbase;
        v8h  w1 = *(const v8h*)(gWbase + 8);
        *(uint4*)&As[0][arow][akoff]     = a0;
        *(uint4*)&As[0][arow][akoff + 8] = a1;
#pragma unroll
        for (int j = 0; j < 8; ++j) {
            Wt[0][wnoff + j][wrow]     = w0[j];
            Wt[0][wnoff + 8 + j][wrow] = w1[j];
        }
    }
    __syncthreads();

    const int nsteps = K >> 5;
    for (int s = 0; s < nsteps; ++s) {
        const int cur = s & 1, nxt = cur ^ 1;
        const bool more = (s + 1 < nsteps);

        // Issue global loads for step s+1 (latency hidden under WMMAs below).
        uint4 a0 = {}, a1 = {};
        v8h  w0 = {}, w1 = {};
        if (more) {
            const int k0 = (s + 1) * 32;
            a0 = *(const uint4*)(gAbase + k0);
            a1 = *(const uint4*)(gAbase + k0 + 8);
            w0 = *(const v8h*)(gWbase + (size_t)k0 * N);
            w1 = *(const v8h*)(gWbase + (size_t)k0 * N + 8);
        }

        // Compute step s from buffer `cur`.
        v16h af[4], bf[2];
#pragma unroll
        for (int mi = 0; mi < 4; ++mi) {
            const int rr = wmb + mi * 16 + mL;
            af[mi] = ld_frag_pair(&As[cur][rr][8 * half], &As[cur][rr][16 + 8 * half]);
        }
#pragma unroll
        for (int ni = 0; ni < 2; ++ni) {
            const int nn = wnb + ni * 16 + mL;
            bf[ni] = ld_frag_pair(&Wt[cur][nn][8 * half], &Wt[cur][nn][16 + 8 * half]);
        }
#pragma unroll
        for (int mi = 0; mi < 4; ++mi)
#pragma unroll
            for (int ni = 0; ni < 2; ++ni)
                acc[mi][ni] = __builtin_amdgcn_wmma_f32_16x16x32_f16(
                    false, af[mi], false, bf[ni], (short)0, acc[mi][ni], false, false);

        // Commit step s+1 into buffer `nxt` (s_wait_loadcnt lands here).
        if (more) {
            *(uint4*)&As[nxt][arow][akoff]     = a0;
            *(uint4*)&As[nxt][arow][akoff + 8] = a1;
#pragma unroll
            for (int j = 0; j < 8; ++j) {
                Wt[nxt][wnoff + j][wrow]     = w0[j];
                Wt[nxt][wnoff + 8 + j][wrow] = w1[j];
            }
        }
        __syncthreads();
    }

    // ---------------- Epilogue (fully compile-time specialized) ----------------
#pragma unroll
    for (int mi = 0; mi < 4; ++mi) {
        const int row0 = M0 + wmb + mi * 16 + 8 * half;   // rows row0 .. row0+7
#pragma unroll
        for (int ni = 0; ni < 2; ++ni) {
            const int col = N0 + wnb + ni * 16 + mL;
            const float bcol = HAS_BIAS ? bias[col] : 0.0f;

            float rv[8];
            if (HAS_RES) {
                const float* rp = residual + (size_t)row0 * N + col;
#pragma unroll
                for (int r = 0; r < 8; ++r) rv[r] = rp[(size_t)r * N];
            }

            float vals[8];
#pragma unroll
            for (int r = 0; r < 8; ++r) {
                float v = acc[mi][ni][r] + bcol;
                if (HAS_RES) v += rv[r];
                if (RELU)    v = fmaxf(v, 0.0f);
                vals[r] = v;
            }

            if (OMODE == 0) {
                float* op = out32 + (size_t)row0 * N + col;
#pragma unroll
                for (int r = 0; r < 8; ++r) op[(size_t)r * N] = vals[r];
            } else if (OMODE == 1) {
                _Float16* op = out16 + (size_t)row0 * N + col;
#pragma unroll
                for (int r = 0; r < 8; ++r) op[(size_t)r * N] = (_Float16)vals[r];
            } else {
                // [B*T, C] -> [B, H, T, D]; a 16-row block never crosses a batch.
                const int bb = row0 >> 11, tt0 = row0 & (SEQ - 1);
                const int hh = col >> 6,  dd = col & 63;
                _Float16* op = out16 +
                    (((size_t)bb * N_HEADS + hh) * SEQ + tt0) * HEAD_SIZE + dd;
#pragma unroll
                for (int r = 0; r < 8; ++r) op[(size_t)r * HEAD_SIZE] = (_Float16)vals[r];
            }
        }
    }
}

// ---------------------------------------------------------------------------
// Flash attention (causal), scale = 1/sqrt(C) = 1/32.
// Grid: (B*H, T/128). Block = 256 threads = 8 waves; wave w handles 16 query
// rows. KV streamed in 32-key blocks via LDS; S and P*V through WMMA.
// ---------------------------------------------------------------------------
__global__ __launch_bounds__(256)
void flash_attn(const _Float16* __restrict__ Qm, const _Float16* __restrict__ Km,
                const _Float16* __restrict__ Vm, _Float16* __restrict__ Om) {
    __shared__ __align__(16) _Float16 Ks[32][64];        // [key][d]
    __shared__ __align__(16) _Float16 Vt[64][40];        // [d][key], pad 8
    __shared__ __align__(16) _Float16 Ps[8][16][40];     // per-wave P tile [m][key]

    const int tid  = threadIdx.x;
    const int lane = tid & 31;
    const int w    = tid >> 5;
    const int mL   = lane & 15;
    const int half = lane >> 4;

    const int bh  = blockIdx.x;            // b*H + h
    const int q0  = blockIdx.y * 128;      // query tile base
    const int qr0 = q0 + w * 16;           // this wave's first query row

    const _Float16* Qb = Qm + (size_t)bh * SEQ * HEAD_SIZE;
    const _Float16* Kb = Km + (size_t)bh * SEQ * HEAD_SIZE;
    const _Float16* Vb = Vm + (size_t)bh * SEQ * HEAD_SIZE;

    // Q fragments for this wave's 16 rows (kept in registers all kernel).
    v16h qf[2];
    {
        const _Float16* qp = Qb + (size_t)(qr0 + mL) * HEAD_SIZE;
#pragma unroll
        for (int c = 0; c < 2; ++c)
            qf[c] = ld_frag_pair(qp + 8 * half + 32 * c, qp + 8 * half + 16 + 32 * c);
    }

    v8f acc[4];
#pragma unroll
    for (int ni = 0; ni < 4; ++ni) acc[ni] = zero8();
    float mrun[8], lrun[8];
#pragma unroll
    for (int r = 0; r < 8; ++r) { mrun[r] = -1e30f; lrun[r] = 0.0f; }

    const float scale = 0.03125f;          // C**-0.5 with C = 1024
    const int nkv = blockIdx.y * 4 + 4;    // 32-key blocks covering keys <= q0+127

    const int srow = tid >> 3, soff = (tid & 7) * 8;   // KV stagers

    for (int blk = 0; blk < nkv; ++blk) {
        const int kv0 = blk * 32;
        // Stage K (as-is) and V (transposed) for this 32-key block.
        {
            uint4 kd = *(const uint4*)(Kb + (size_t)(kv0 + srow) * HEAD_SIZE + soff);
            *(uint4*)&Ks[srow][soff] = kd;
            v8h vd = *(const v8h*)(Vb + (size_t)(kv0 + srow) * HEAD_SIZE + soff);
#pragma unroll
            for (int j = 0; j < 8; ++j) Vt[soff + j][srow] = vd[j];
        }
        // Prefetch next KV block (WGP scope -> near caches) while consuming this one.
        if (blk + 1 < nkv) {
            __builtin_prefetch(Kb + (size_t)(kv0 + 32 + srow) * HEAD_SIZE + soff, 0, 3);
            __builtin_prefetch(Vb + (size_t)(kv0 + 32 + srow) * HEAD_SIZE + soff, 0, 3);
        }
        __syncthreads();

        // Wave-uniform causal skip; readfirstlane makes it a provable scalar
        // branch so EXEC stays all-ones around the WMMAs (ISA requirement).
        const int doblk = __builtin_amdgcn_readfirstlane((kv0 <= qr0 + 15) ? 1 : 0);
        if (doblk) {
            // S = Q K^T (two 16x16 tiles over the 32 keys)
            v8f S[2];
#pragma unroll
            for (int nt = 0; nt < 2; ++nt) {
                const int n = nt * 16 + mL;   // key lane
                v8f s = zero8();
#pragma unroll
                for (int c = 0; c < 2; ++c) {
                    v16h kf = ld_frag_pair(&Ks[n][8 * half + 32 * c],
                                           &Ks[n][16 + 8 * half + 32 * c]);
                    s = __builtin_amdgcn_wmma_f32_16x16x32_f16(
                        false, qf[c], false, kf, (short)0, s, false, false);
                }
                S[nt] = s;
            }
            // Online softmax (per row r+8*half; 16-lane reductions)
#pragma unroll
            for (int r = 0; r < 8; ++r) {
                const int rowt = qr0 + r + 8 * half;
                float s0 = S[0][r] * scale;
                float s1 = S[1][r] * scale;
                if (kv0 + mL > rowt)      s0 = -1e30f;
                if (kv0 + 16 + mL > rowt) s1 = -1e30f;
                float rm = fmaxf(s0, s1);
#pragma unroll
                for (int off = 8; off > 0; off >>= 1) rm = fmaxf(rm, __shfl_xor(rm, off, 16));
                const float mnew = fmaxf(mrun[r], rm);
                const float sc   = __expf(mrun[r] - mnew);
                const float p0   = __expf(s0 - mnew);
                const float p1   = __expf(s1 - mnew);
                float rs = p0 + p1;
#pragma unroll
                for (int off = 8; off > 0; off >>= 1) rs += __shfl_xor(rs, off, 16);
                lrun[r] = lrun[r] * sc + rs;
                mrun[r] = mnew;
#pragma unroll
                for (int ni = 0; ni < 4; ++ni) acc[ni][r] *= sc;
                Ps[w][r + 8 * half][mL]      = (_Float16)p0;
                Ps[w][r + 8 * half][16 + mL] = (_Float16)p1;
            }
            // P (A-frag) x V (B-frags): acc += P @ V. DS ops are in-order
            // within a wave, so no barrier needed for the per-wave Ps tile.
            v16h pf = ld_frag_pair(&Ps[w][mL][8 * half], &Ps[w][mL][16 + 8 * half]);
#pragma unroll
            for (int ni = 0; ni < 4; ++ni) {
                const int d = ni * 16 + mL;
                v16h vf = ld_frag_pair(&Vt[d][8 * half], &Vt[d][16 + 8 * half]);
                acc[ni] = __builtin_amdgcn_wmma_f32_16x16x32_f16(
                    false, pf, false, vf, (short)0, acc[ni], false, false);
            }
        }
        __syncthreads();
    }

    // Write O back as [B, T, C] (heads concatenated) in f16 for proj GEMM.
    const int bb = bh / N_HEADS, hh = bh % N_HEADS;
    _Float16* ob = Om + ((size_t)bb * SEQ + qr0 + 8 * half) * N_EMBD + hh * HEAD_SIZE + mL;
#pragma unroll
    for (int r = 0; r < 8; ++r) {
        const float inv = 1.0f / lrun[r];
#pragma unroll
        for (int ni = 0; ni < 4; ++ni) {
            ob[(size_t)r * N_EMBD + ni * 16] = (_Float16)(acc[ni][r] * inv);
        }
    }
}

// ---------------------------------------------------------------------------
// Host-side launcher
// ---------------------------------------------------------------------------
extern "C" void kernel_launch(void* const* d_in, const int* in_sizes, int n_in,
                              void* d_out, int out_size, void* d_ws, size_t ws_size,
                              hipStream_t stream) {
    const float* x     = (const float*)d_in[0];
    const float* Wq    = (const float*)d_in[1];
    const float* Wk    = (const float*)d_in[2];
    const float* Wv    = (const float*)d_in[3];
    const float* Wproj = (const float*)d_in[4];
    const float* bproj = (const float*)d_in[5];
    const float* W1    = (const float*)d_in[6];
    const float* b1    = (const float*)d_in[7];
    const float* W2    = (const float*)d_in[8];
    const float* b2    = (const float*)d_in[9];
    const float* ln1w  = (const float*)d_in[10];
    const float* ln1b  = (const float*)d_in[11];
    const float* ln2w  = (const float*)d_in[12];
    const float* ln2b  = (const float*)d_in[13];

    // Workspace carve-up (all 256B-aligned by construction)
    char* ws = (char*)d_ws;
    size_t off = 0;
    auto take = [&](size_t bytes) { char* p = ws + off; off += (bytes + 255) & ~(size_t)255; return p; };
    _Float16* h16   = (_Float16*)take((size_t)ROWS * N_EMBD * 2);
    _Float16* q16   = (_Float16*)take((size_t)ROWS * N_EMBD * 2);
    _Float16* k16   = (_Float16*)take((size_t)ROWS * N_EMBD * 2);
    _Float16* v16   = (_Float16*)take((size_t)ROWS * N_EMBD * 2);
    _Float16* attn16= (_Float16*)take((size_t)ROWS * N_EMBD * 2);
    float*    x1    = (float*)   take((size_t)ROWS * N_EMBD * 4);
    _Float16* h2_16 = (_Float16*)take((size_t)ROWS * N_EMBD * 2);
    _Float16* ff16  = (_Float16*)take((size_t)ROWS * FFN * 2);
    _Float16* wq16  = (_Float16*)take((size_t)N_EMBD * N_EMBD * 2);
    _Float16* wk16  = (_Float16*)take((size_t)N_EMBD * N_EMBD * 2);
    _Float16* wv16  = (_Float16*)take((size_t)N_EMBD * N_EMBD * 2);
    _Float16* wp16  = (_Float16*)take((size_t)N_EMBD * N_EMBD * 2);
    _Float16* w116  = (_Float16*)take((size_t)N_EMBD * FFN * 2);
    _Float16* w216  = (_Float16*)take((size_t)FFN * N_EMBD * 2);

    const int nW = N_EMBD * N_EMBD;
    const int nF = N_EMBD * FFN;
    cast_f32_to_f16<<<1024, 256, 0, stream>>>(Wq, wq16, nW);
    cast_f32_to_f16<<<1024, 256, 0, stream>>>(Wk, wk16, nW);
    cast_f32_to_f16<<<1024, 256, 0, stream>>>(Wv, wv16, nW);
    cast_f32_to_f16<<<1024, 256, 0, stream>>>(Wproj, wp16, nW);
    cast_f32_to_f16<<<2048, 256, 0, stream>>>(W1, w116, nF);
    cast_f32_to_f16<<<2048, 256, 0, stream>>>(W2, w216, nF);

    // LN1
    ln_kernel<<<ROWS, 256, 0, stream>>>(x, ln1w, ln1b, h16);

    // QKV projections ([B,H,T,D] f16 outputs)
    dim3 gQKV(N_EMBD / 128, ROWS / 128);
    gemm_wmma<N_EMBD, false, false, false, 2><<<gQKV, 256, 0, stream>>>(
        h16, wq16, nullptr, nullptr, nullptr, q16, ROWS, N_EMBD);
    gemm_wmma<N_EMBD, false, false, false, 2><<<gQKV, 256, 0, stream>>>(
        h16, wk16, nullptr, nullptr, nullptr, k16, ROWS, N_EMBD);
    gemm_wmma<N_EMBD, false, false, false, 2><<<gQKV, 256, 0, stream>>>(
        h16, wv16, nullptr, nullptr, nullptr, v16, ROWS, N_EMBD);

    // Causal flash attention
    flash_attn<<<dim3(BATCH * N_HEADS, SEQ / 128), 256, 0, stream>>>(q16, k16, v16, attn16);

    // Output projection + bias + residual -> x1 (f32)
    gemm_wmma<N_EMBD, true, true, false, 0><<<gQKV, 256, 0, stream>>>(
        attn16, wp16, bproj, x, x1, nullptr, ROWS, N_EMBD);

    // LN2
    ln_kernel<<<ROWS, 256, 0, stream>>>(x1, ln2w, ln2b, h2_16);

    // FFN1: relu(h2 @ W1 + b1) -> f16
    gemm_wmma<FFN, true, false, true, 1><<<dim3(FFN / 128, ROWS / 128), 256, 0, stream>>>(
        h2_16, w116, b1, nullptr, nullptr, ff16, ROWS, N_EMBD);

    // FFN2: x1 + ff @ W2 + b2 -> d_out (f32)
    gemm_wmma<N_EMBD, true, true, false, 0><<<gQKV, 256, 0, stream>>>(
        ff16, w216, b2, x1, (float*)d_out, nullptr, ROWS, FFN);
}
